// FactorAtt_ConvRelPosEnc_21930103014121
// MI455X (gfx1250) — matrix operand order repaired and checked by reading, not compiled, not run on hardware
//
#include <hip/hip_runtime.h>

// FactorAtt + ConvRelPosEnc, MI455X (gfx1250, wave32, WMMA bf16->f32 accumulate)
// B=32, N=1024 (32x32), C=768, heads=8, Ch=96

typedef __attribute__((ext_vector_type(16))) __bf16 v16bf;
typedef __attribute__((ext_vector_type(8)))  __bf16 bf8;    // 16 bytes
typedef __attribute__((ext_vector_type(4)))  __bf16 bf4;    // 8 bytes
typedef __attribute__((ext_vector_type(8)))  float  v8f;
typedef __attribute__((ext_vector_type(4)))  float  f4;
typedef __attribute__((ext_vector_type(4)))  int    v4i;    // 16 bytes

union Frag { v16bf v; bf8 h[2]; };

// ---- async global->LDS copy (gfx1250), probed with compile-safe fallback ----
#if defined(__gfx1250__) && \
    __has_builtin(__builtin_amdgcn_global_load_async_to_lds_b128) && \
    __has_builtin(__builtin_amdgcn_s_wait_asynccnt)
#define ASYNC_LDS 1
typedef __attribute__((address_space(1))) v4i* as1_v4i_p;
typedef __attribute__((address_space(3))) v4i* as3_v4i_p;
__device__ __forceinline__ void copy16(const void* g, void* l) {
  __builtin_amdgcn_global_load_async_to_lds_b128(
      (as1_v4i_p)g, (as3_v4i_p)l, 0, 0);
}
#define LDS_COPY_FENCE() __builtin_amdgcn_s_wait_asynccnt(0)
#else
#define ASYNC_LDS 0
__device__ __forceinline__ void copy16(const void* g, void* l) {
  *(bf8*)l = *(const bf8*)g;
}
#define LDS_COPY_FENCE() ((void)0)
#endif

// A-operand (16-bit 16x32 MxK): lanes 0-15 hold K={o..o+7, o+16..o+23}, o=0;
// lanes 16-31: o=8.  rowbase points at lds[row*stride + kb].
__device__ __forceinline__ v16bf frag_a(const __bf16* rowbase, int lane) {
  int o = (lane < 16) ? 0 : 8;
  Frag f;
  f.h[0] = *(const bf8*)(rowbase + o);
  f.h[1] = *(const bf8*)(rowbase + o + 16);
  return f.v;
}

// B-operand (16-bit 32x16 KxN): lane = column n%16; lanes 0-15 hold
// K={o..o+7, o+16..o+23}, o=0; lanes 16-31: o=8 (matches frag_a).
__device__ __forceinline__ v16bf frag_b(const __bf16* rowbase, int lane) {
  int o = (lane < 16) ? 0 : 8;
  Frag f;
  f.h[0] = *(const bf8*)(rowbase + o);
  f.h[1] = *(const bf8*)(rowbase + o + 16);
  return f.v;
}

// ---------------------------------------------------------------------------
// Kernel 0: one-shot fp32 -> bf16 conversion of x and w_qkv (native casts).
// ---------------------------------------------------------------------------
__global__ __launch_bounds__(256) void to_bf16(const float* __restrict__ x,
                                               const float* __restrict__ w,
                                               __bf16* __restrict__ xb,
                                               __bf16* __restrict__ wb) {
  size_t i = ((size_t)blockIdx.x * 256 + threadIdx.x) * 4;
  const size_t NX = (size_t)32768 * 768;      // 25165824
  if (i < NX) {
    f4 a = *(const f4*)(x + i);
    bf4 t = { (__bf16)a.x, (__bf16)a.y, (__bf16)a.z, (__bf16)a.w };
    *(bf4*)(xb + i) = t;
  } else {
    size_t j = i - NX;                        // < 2304*768
    f4 a = *(const f4*)(w + j);
    bf4 t = { (__bf16)a.x, (__bf16)a.y, (__bf16)a.z, (__bf16)a.w };
    *(bf4*)(wb + j) = t;
  }
}

// ---------------------------------------------------------------------------
// Kernel 1: qkv = x @ w_qkv^T ; scatter fp32 into q/k/v [B,h,N,Ch] (+ q bf16).
// M=32768, N=2304, K=768.  Block = 128x128 tile, 8 waves (4x2), BK=32.
// Double-buffered async pipeline: one barrier per K-step, copies overlap WMMA.
// ---------------------------------------------------------------------------
__global__ __launch_bounds__(256) void qkv_gemm(const __bf16* __restrict__ xb,
                                                const __bf16* __restrict__ wb,
                                                float* __restrict__ qw,
                                                float* __restrict__ kw,
                                                float* __restrict__ vw,
                                                __bf16* __restrict__ qb) {
  __shared__ __align__(16) __bf16 lsa[2][128 * 40];
  __shared__ __align__(16) __bf16 lsb[2][128 * 40];
  const int tid  = threadIdx.x;
  const int lane = tid & 31;
  const int wv   = tid >> 5;
  const int n0   = blockIdx.x * 128;   // 0..2176
  const int m0   = blockIdx.y * 128;   // 0..32640
  const int wm   = wv & 3;             // 32-row strip
  const int wn   = wv >> 2;            // 64-col strip

  auto stage = [&](int buf, int k0) {
    #pragma unroll
    for (int it = 0; it < 2; it++) {   // 512 16B slots per matrix
      int li  = tid + it * 256;
      int row = li >> 2;
      int sg  = (li & 3) * 8;
      copy16(xb + (size_t)(m0 + row) * 768 + k0 + sg, &lsa[buf][row * 40 + sg]);
      copy16(wb + (size_t)(n0 + row) * 768 + k0 + sg, &lsb[buf][row * 40 + sg]);
    }
  };

  stage(0, 0);                         // prologue

  v8f acc[2][4] = {};

  for (int k0 = 0; k0 < 768; k0 += 32) {
    const int buf = (k0 >> 5) & 1;
    // Wait for own stage-n copies; barrier makes all waves' stage-n LDS writes
    // visible AND confirms everyone finished reading stage n-1's buffer.
    LDS_COPY_FENCE();
    __syncthreads();
    if (k0 + 32 < 768) stage(buf ^ 1, k0 + 32);   // overlap with WMMA below

    v16bf afr[2], bfr[4];
    #pragma unroll
    for (int i = 0; i < 2; i++) {
      int row = wm * 32 + i * 16 + (lane & 15);
      afr[i] = frag_a(&lsa[buf][row * 40], lane);
    }
    #pragma unroll
    for (int j = 0; j < 4; j++) {
      int row = wn * 64 + j * 16 + (lane & 15);
      bfr[j] = frag_b(&lsb[buf][row * 40], lane);
    }
    #pragma unroll
    for (int i = 0; i < 2; i++)
      #pragma unroll
      for (int j = 0; j < 4; j++)
        acc[i][j] = __builtin_amdgcn_wmma_f32_16x16x32_bf16(
            false, afr[i], false, bfr[j], (short)0, acc[i][j], false, false);
  }

  // Epilogue: split columns into q/k/v, head-major [B,h,N,Ch]; q also in bf16.
  #pragma unroll
  for (int i = 0; i < 2; i++) {
    #pragma unroll
    for (int j = 0; j < 4; j++) {
      int gn    = n0 + wn * 64 + j * 16 + (lane & 15);
      int which = gn / 768;            // uniform across the 16-wide tile
      int rem   = gn - which * 768;
      int hd    = rem / 96;
      int ch    = rem - hd * 96;
      float* base = (which == 0) ? qw : (which == 1) ? kw : vw;
      #pragma unroll
      for (int r = 0; r < 8; r++) {
        int gm = m0 + wm * 32 + i * 16 + r + ((lane >> 4) << 3);
        int b  = gm >> 10;
        int n  = gm & 1023;
        size_t idx = (size_t)((b * 8 + hd) * 1024 + n) * 96 + ch;
        base[idx] = acc[i][j][r];
        if (which == 0) qb[idx] = (__bf16)acc[i][j][r];
      }
    }
  }
}

// ---------------------------------------------------------------------------
// Kernel 2: per (b,h): online softmax stats over tokens, then
// kvT[cv,ck] = (Ch^-0.5 / sum_ck) * sum_n exp(k[n,ck]-max_ck) * v[n,cv]  (bf16)
// ---------------------------------------------------------------------------
__global__ __launch_bounds__(256) void kv_kernel(const float* __restrict__ kw,
                                                 const float* __restrict__ vw,
                                                 __bf16* __restrict__ kvT) {
  __shared__ float cmax[96];
  __shared__ float cscale[96];
  __shared__ __align__(16) __bf16 lsk[96 * 40];
  __shared__ __align__(16) __bf16 lsv[96 * 40];
  const int tid  = threadIdx.x;
  const int lane = tid & 31;
  const int wv   = tid >> 5;
  const int bh   = blockIdx.x;
  const float* kbase = kw + (size_t)bh * 1024 * 96;
  const float* vbase = vw + (size_t)bh * 1024 * 96;

  if (tid < 96) {                       // online (max,sumexp) per column
    float m = -3.4e38f, s = 0.f;
    for (int n = 0; n < 1024; n++) {
      float xv = kbase[(size_t)n * 96 + tid];
      float nm = fmaxf(m, xv);
      s = s * __expf(m - nm) + __expf(xv - nm);
      m = nm;
    }
    cmax[tid]   = m;
    cscale[tid] = 0.1020620726159658f / s;   // 96^-0.5 folded in
  }
  __syncthreads();

  v8f acc[6] = {};
  for (int c = 0; c < 32; c++) {
    const int nb = c * 32;
    __syncthreads();
    #pragma unroll
    for (int it = 0; it < 12; it++) {   // 32x96 chunk, transposed into LDS
      int i  = tid + it * 256;
      int nl = i / 96;
      int ch = i - nl * 96;
      float kx = __expf(kbase[(size_t)(nb + nl) * 96 + ch] - cmax[ch]);
      lsk[ch * 40 + nl] = (__bf16)kx;
      lsv[ch * 40 + nl] = (__bf16)vbase[(size_t)(nb + nl) * 96 + ch];
    }
    __syncthreads();
    if (wv < 6) {
      int mrow = wv * 16 + (lane & 15);
      v16bf afr = frag_a(&lsk[mrow * 40], lane);
      #pragma unroll
      for (int j = 0; j < 6; j++) {
        int nrow = j * 16 + (lane & 15);
        v16bf bfr = frag_b(&lsv[nrow * 40], lane);
        acc[j] = __builtin_amdgcn_wmma_f32_16x16x32_bf16(
            false, afr, false, bfr, (short)0, acc[j], false, false);
      }
    }
  }
  if (wv < 6) {
    __bf16* outp = kvT + (size_t)bh * 96 * 96;
    #pragma unroll
    for (int j = 0; j < 6; j++) {
      #pragma unroll
      for (int r = 0; r < 8; r++) {
        int ck = wv * 16 + r + ((lane >> 4) << 3);
        int cv = j * 16 + (lane & 15);
        outp[cv * 96 + ck] = (__bf16)(acc[j][r] * cscale[ck]);  // transposed
      }
    }
  }
}

// ---------------------------------------------------------------------------
// Kernel 3: depthwise 3x3 SAME conv over v in [B,h,(H,W),Ch] layout.
// ---------------------------------------------------------------------------
__global__ __launch_bounds__(256) void crpe_conv(const float* __restrict__ vw,
                                                 const float* __restrict__ cw,
                                                 const float* __restrict__ cb,
                                                 float* __restrict__ vconv) {
  size_t i = (size_t)blockIdx.x * 256 + threadIdx.x;   // 32*8*1024*96 total
  int ch = (int)(i % 96);
  size_t t = i / 96;
  int n  = (int)(t % 1024);
  int bh = (int)(t / 1024);
  int y = n >> 5, xx = n & 31;
  int c = (bh & 7) * 96 + ch;
  const float* wp = cw + c * 9;
  const float* vb = vw + (size_t)bh * 1024 * 96;
  float s = cb[c];
  #pragma unroll
  for (int ky = 0; ky < 3; ky++) {
    int iy = y + ky - 1;
    if (iy < 0 || iy > 31) continue;
    #pragma unroll
    for (int kx = 0; kx < 3; kx++) {
      int ix = xx + kx - 1;
      if (ix < 0 || ix > 31) continue;
      s += wp[ky * 3 + kx] * vb[(size_t)(iy * 32 + ix) * 96 + ch];
    }
  }
  vconv[i] = s;
}

// ---------------------------------------------------------------------------
// Kernel 4: per (b,h), 128-row chunk: factor_att = q @ kv (WMMA),
// out[b,n,h*96+ch] = factor_att + q * v_back   (scale already folded into kv)
// ---------------------------------------------------------------------------
__global__ __launch_bounds__(256) void out_kernel(const float* __restrict__ qw,
                                                  const __bf16* __restrict__ qb,
                                                  const __bf16* __restrict__ kvT,
                                                  const float* __restrict__ vconv,
                                                  float* __restrict__ out) {
  __shared__ __align__(16) __bf16 lsq[128 * 104];
  __shared__ __align__(16) __bf16 lskv[96 * 104];
  const int tid  = threadIdx.x;
  const int lane = tid & 31;
  const int wv   = tid >> 5;
  const int bh   = blockIdx.x >> 3;
  const int n0   = (blockIdx.x & 7) * 128;
  const int b    = bh >> 3, hd = bh & 7;

  const __bf16* qgb = qb + ((size_t)bh * 1024 + n0) * 96;
  #pragma unroll
  for (int it = 0; it < 6; it++) {      // q chunk 128x96 bf16: 1536 16B slots
    int li = tid + it * 256;
    int r = li / 12, sg = (li % 12) * 8;
    copy16(qgb + (size_t)r * 96 + sg, &lsq[r * 104 + sg]);
  }
  const __bf16* kvb = kvT + (size_t)bh * 96 * 96;
  #pragma unroll
  for (int it = 0; it < 5; it++) {      // kvT 96x96 bf16: 1152 16B slots
    int li = tid + it * 256;
    if (li < 1152) {
      int r = li / 12, sg = (li % 12) * 8;
      copy16(kvb + r * 96 + sg, &lskv[r * 104 + sg]);
    }
  }
  LDS_COPY_FENCE();
  __syncthreads();

  v8f acc[6] = {};
  const int m0 = wv * 16;
  #pragma unroll
  for (int kb = 0; kb < 96; kb += 32) {
    int mrow = m0 + (lane & 15);
    v16bf afr = frag_a(&lsq[mrow * 104 + kb], lane);
    #pragma unroll
    for (int j = 0; j < 6; j++) {
      int nrow = j * 16 + (lane & 15);
      v16bf bfr = frag_b(&lskv[nrow * 104 + kb], lane);
      acc[j] = __builtin_amdgcn_wmma_f32_16x16x32_bf16(
          false, afr, false, bfr, (short)0, acc[j], false, false);
    }
  }

  #pragma unroll
  for (int j = 0; j < 6; j++) {
    #pragma unroll
    for (int r = 0; r < 8; r++) {
      int n  = n0 + m0 + r + ((lane >> 4) << 3);
      int cv = j * 16 + (lane & 15);
      size_t qi = ((size_t)bh * 1024 + n) * 96 + cv;
      float crpe = qw[qi] * vconv[qi];
      out[((size_t)(b * 1024 + n)) * 768 + hd * 96 + cv] = acc[j][r] + crpe;
    }
  }
}

// ---------------------------------------------------------------------------
extern "C" void kernel_launch(void* const* d_in, const int* in_sizes, int n_in,
                              void* d_out, int out_size, void* d_ws, size_t ws_size,
                              hipStream_t stream) {
  (void)in_sizes; (void)n_in; (void)out_size; (void)ws_size;
  const float* x  = (const float*)d_in[0];   // [32,1024,768]
  const float* w  = (const float*)d_in[1];   // [2304,768]
  const float* cw = (const float*)d_in[2];   // [768,1,3,3]
  const float* cb = (const float*)d_in[3];   // [768]
  float* out = (float*)d_out;                // [32,1024,768]

  char* ws = (char*)d_ws;
  const size_t per = (size_t)32 * 8 * 1024 * 96 * sizeof(float);   // 96 MB
  float*  qw    = (float*)(ws);
  float*  kw    = (float*)(ws + per);
  float*  vw    = (float*)(ws + 2 * per);
  float*  vconv = (float*)(ws + 3 * per);
  size_t  off   = 4 * per;
  __bf16* kvT   = (__bf16*)(ws + off);  off += (size_t)32*8*96*96*2;     // 4.5 MB
  __bf16* xb    = (__bf16*)(ws + off);  off += (size_t)32768*768*2;      // 48 MB
  __bf16* wb    = (__bf16*)(ws + off);  off += (size_t)2304*768*2;       // 3.4 MB
  __bf16* qb    = (__bf16*)(ws + off);                                   // 48 MB

  to_bf16  <<<26304, 256, 0, stream>>>(x, w, xb, wb);
  dim3 g1(18, 256);                                  // 2304/128 x 32768/128
  qkv_gemm <<<g1,    256, 0, stream>>>(xb, wb, qw, kw, vw, qb);
  kv_kernel<<<256,   256, 0, stream>>>(kw, vw, kvT); // one block per (b,h)
  crpe_conv<<<98304, 256, 0, stream>>>(vw, cw, cb, vconv);
  out_kernel<<<2048, 256, 0, stream>>>(qw, qb, kvT, vconv, out);
}
